// EncoderBlock_8787503087716
// MI455X (gfx1250) — matrix operand order repaired and checked by reading, not compiled, not run on hardware
//
#include <hip/hip_runtime.h>
#include <math.h>

// ---------------------------------------------------------------------------
// Transformer encoder block for MI455X (gfx1250, wave32, WMMA).
// bf16 activations/weights + f32 accumulation via v_wmma_f32_16x16x32_bf16.
// Fragment operands are ds_load_b128 pairs; global->LDS staging uses CDNA5
// GLOBAL_LOAD_ASYNC_TO_LDS_B128 with a double-buffered k-loop in the GEMM.
// ---------------------------------------------------------------------------

typedef __attribute__((ext_vector_type(16))) __bf16        v16bf;
typedef __attribute__((ext_vector_type(8)))  float         v8f;
typedef __attribute__((ext_vector_type(4)))  unsigned int  v4u;
typedef __attribute__((ext_vector_type(8)))  unsigned int  v8u;

#define D_MODEL 1024
#define NSEQ    1024
#define BATCH   8
#define NHEAD   16
#define DHEAD   64
#define HIDDEN  4096
#define MTOT    (BATCH * NSEQ)

__device__ __forceinline__ unsigned short f32_to_bf16(float f) {
  unsigned int u = __builtin_bit_cast(unsigned int, f);
  u += 0x7FFFu + ((u >> 16) & 1u);  // round-to-nearest-even
  return (unsigned short)(u >> 16);
}

// Two 16-byte LDS loads -> one 32-byte bf16 fragment (ds_load_b128 x2).
__device__ __forceinline__ v16bf frag2(const unsigned short* p0,
                                       const unsigned short* p1) {
  v4u lo = *(const v4u*)p0;
  v4u hi = *(const v4u*)p1;
  v8u c = __builtin_shufflevector(lo, hi, 0, 1, 2, 3, 4, 5, 6, 7);
  return __builtin_bit_cast(v16bf, c);
}

// CDNA5 async copy: 16 bytes/lane global -> LDS, tracked by ASYNCcnt.
__device__ __forceinline__ void async_cp_b128(unsigned lds_addr,
                                              const unsigned short* g) {
  asm volatile("global_load_async_to_lds_b128 %0, %1, off"
               :: "v"(lds_addr), "v"(g)
               : "memory");
}
__device__ __forceinline__ void async_wait0() {
  asm volatile("s_wait_asynccnt 0" ::: "memory");
}
__device__ __forceinline__ unsigned lds_addr_of(const void* p) {
  return (unsigned)(unsigned long long)p;  // LDS aperture: low 32 bits
}

// ---------------------------------------------------------------------------
// fp32 -> bf16 streaming convert (weights)
// ---------------------------------------------------------------------------
__global__ void cvt_f32_bf16(const float* __restrict__ in,
                             unsigned short* __restrict__ out, int n) {
  for (int i = blockIdx.x * blockDim.x + threadIdx.x; i < n;
       i += gridDim.x * blockDim.x)
    out[i] = f32_to_bf16(in[i]);
}

// ---------------------------------------------------------------------------
// LayerNorm: one 256-thread block per row of 1024, bf16 output.
// float4 loads, packed 4xbf16 (64-bit) stores.
// ---------------------------------------------------------------------------
__global__ __launch_bounds__(256) void ln_bf16_kernel(
    const float* __restrict__ x, const float* __restrict__ g,
    const float* __restrict__ b, unsigned short* __restrict__ out) {
  const int row = blockIdx.x;
  const int tid = threadIdx.x;
  const float4 v = ((const float4*)(x + (size_t)row * D_MODEL))[tid];
  __shared__ float red[256];
  red[tid] = v.x + v.y + v.z + v.w;
  __syncthreads();
  for (int w = 128; w > 0; w >>= 1) {
    if (tid < w) red[tid] += red[tid + w];
    __syncthreads();
  }
  const float mu = red[0] * (1.f / D_MODEL);
  __syncthreads();
  float dx = v.x - mu, dy = v.y - mu, dz = v.z - mu, dw = v.w - mu;
  red[tid] = dx * dx + dy * dy + dz * dz + dw * dw;
  __syncthreads();
  for (int w = 128; w > 0; w >>= 1) {
    if (tid < w) red[tid] += red[tid + w];
    __syncthreads();
  }
  const float rs = rsqrtf(red[0] * (1.f / D_MODEL) + 1e-5f);
  const float4 gg = ((const float4*)g)[tid];
  const float4 bb = ((const float4*)b)[tid];
  unsigned long long pk =
      (unsigned long long)f32_to_bf16(dx * rs * gg.x + bb.x) |
      ((unsigned long long)f32_to_bf16(dy * rs * gg.y + bb.y) << 16) |
      ((unsigned long long)f32_to_bf16(dz * rs * gg.z + bb.z) << 32) |
      ((unsigned long long)f32_to_bf16(dw * rs * gg.w + bb.w) << 48);
  ((unsigned long long*)(out + (size_t)row * D_MODEL))[tid] = pk;
}

// ---------------------------------------------------------------------------
// bf16 WMMA GEMM: C[M,Nc] = A[M,K] * W[K,Nc] (+bias, +epilogue)
// Block tile 128x64, BK=32, 8 waves; wave = 32x32 (2x2 WMMA per K-step).
// Double-buffered: async A-tile copies (global_load_async_to_lds_b128),
// register-staged transposed B tile; one barrier per K-step.
// MODE 0: bf16 store; MODE 1: exact GELU -> bf16; MODE 2: +resid -> f32.
// ---------------------------------------------------------------------------
template <int MODE>
__global__ __launch_bounds__(256) void gemm_bf16(
    const unsigned short* __restrict__ A, const unsigned short* __restrict__ Bw,
    const float* __restrict__ bias, const float* __restrict__ resid,
    float* __restrict__ outF, unsigned short* __restrict__ outB,
    int M, int Nc, int K) {
  constexpr int BM = 128, BN = 64, BK = 32;
  constexpr int LDA = BK + 8;   // 40 elems = 80B rows (16B aligned)
  constexpr int LDBT = BK + 8;  // transposed B: [n][k]
  __shared__ unsigned short sA[2][BM * LDA];
  __shared__ unsigned short sBt[2][BN * LDBT];
  const int tid = threadIdx.x;
  const int lane = tid & 31, wave = tid >> 5;
  const int wm = wave & 3, wn = wave >> 2;
  const int nl = lane & 15, half = lane >> 4;
  const int bm = blockIdx.y * BM, bn = blockIdx.x * BN;

  v8f acc[2][2];
#pragma unroll
  for (int mi = 0; mi < 2; mi++)
#pragma unroll
    for (int ni = 0; ni < 2; ni++)
#pragma unroll
      for (int i = 0; i < 8; i++) acc[mi][ni][i] = 0.f;

  // Per-thread staging geometry (advance pointers by constants per K-step)
  const int a_ml = tid >> 2, a_kc = tid & 3;          // A: 2 chunks of 8 bf16
  const int b_kl = tid >> 3, b_nc = tid & 7;          // B: 1 chunk of 8 bf16
  const unsigned short* gA = A + (size_t)(bm + a_ml) * K + a_kc * 8;
  const unsigned short* gB = Bw + (size_t)b_kl * Nc + bn + b_nc * 8;
  const size_t gA2off = (size_t)64 * K;               // second A chunk (+64 rows)
  const unsigned lA[2] = {
      lds_addr_of(&sA[0][a_ml * LDA + a_kc * 8]),
      lds_addr_of(&sA[1][a_ml * LDA + a_kc * 8])};
  const unsigned lA2 = 64 * LDA * 2;                  // +64 rows, bytes

  // Fragment offsets (constant per thread)
  const int aoff0 = (wm * 32 + nl) * LDA + 16 * half;
  const int aoff1 = aoff0 + 16 * LDA;
  const int boff0 = (wn * 32 + nl) * LDBT + 16 * half;
  const int boff1 = boff0 + 16 * LDBT;

  const int nk = K / BK;  // even for all our shapes

  // Prologue: stage tile 0 into buffer 0
  async_cp_b128(lA[0], gA);
  async_cp_b128(lA[0] + lA2, gA + gA2off);
  {
    v4u br = *(const v4u*)gB;
#pragma unroll
    for (int e = 0; e < 4; e++) {
      unsigned w = br[e];
      sBt[0][(b_nc * 8 + 2 * e) * LDBT + b_kl] = (unsigned short)(w & 0xffffu);
      sBt[0][(b_nc * 8 + 2 * e + 1) * LDBT + b_kl] = (unsigned short)(w >> 16);
    }
  }
  gA += BK;
  gB += (size_t)BK * Nc;
  async_wait0();
  __syncthreads();

  auto step = [&](const unsigned short* cA, const unsigned short* cB,
                  unsigned short* nB, unsigned nAaddr, bool pre) {
    v4u br;
    if (pre) {
      async_cp_b128(nAaddr, gA);
      async_cp_b128(nAaddr + lA2, gA + gA2off);
      br = *(const v4u*)gB;  // overlaps with WMMAs below
      gA += BK;
      gB += (size_t)BK * Nc;
    }
    v16bf af0 = frag2(cA + aoff0, cA + aoff0 + 8);
    v16bf af1 = frag2(cA + aoff1, cA + aoff1 + 8);
    v16bf bf0 = frag2(cB + boff0, cB + boff0 + 8);
    v16bf bf1 = frag2(cB + boff1, cB + boff1 + 8);
    acc[0][0] = __builtin_amdgcn_wmma_f32_16x16x32_bf16(
        false, af0, false, bf0, (short)0, acc[0][0], false, false);
    acc[0][1] = __builtin_amdgcn_wmma_f32_16x16x32_bf16(
        false, af0, false, bf1, (short)0, acc[0][1], false, false);
    acc[1][0] = __builtin_amdgcn_wmma_f32_16x16x32_bf16(
        false, af1, false, bf0, (short)0, acc[1][0], false, false);
    acc[1][1] = __builtin_amdgcn_wmma_f32_16x16x32_bf16(
        false, af1, false, bf1, (short)0, acc[1][1], false, false);
    if (pre) {
#pragma unroll
      for (int e = 0; e < 4; e++) {
        unsigned w = br[e];
        nB[(b_nc * 8 + 2 * e) * LDBT + b_kl] = (unsigned short)(w & 0xffffu);
        nB[(b_nc * 8 + 2 * e + 1) * LDBT + b_kl] = (unsigned short)(w >> 16);
      }
      async_wait0();
    }
    __syncthreads();
  };

  for (int kb = 0; kb < nk; kb += 2) {
    step(sA[0], sBt[0], sBt[1], lA[1], true);
    step(sA[1], sBt[1], sBt[0], lA[0], kb + 2 < nk);
  }

  // Epilogue (C layout: vgpr r -> row r + 8*half, lane -> col nl)
#pragma unroll
  for (int mi = 0; mi < 2; mi++) {
#pragma unroll
    for (int ni = 0; ni < 2; ni++) {
      int col = bn + wn * 32 + ni * 16 + nl;
      float bc = bias[col];
#pragma unroll
      for (int r = 0; r < 8; r++) {
        int row = bm + wm * 32 + mi * 16 + r + 8 * half;
        float v = acc[mi][ni][r] + bc;
        size_t idx = (size_t)row * Nc + col;
        if (MODE == 0) {
          outB[idx] = f32_to_bf16(v);
        } else if (MODE == 1) {
          float ge = 0.5f * v * (1.f + erff(v * 0.70710678118654752f));
          outB[idx] = f32_to_bf16(ge);
        } else {
          outF[idx] = v + resid[idx];
        }
      }
    }
  }
}

// ---------------------------------------------------------------------------
// Flash-style attention: block = (64 query rows) x (one b,h); 8 waves,
// wave = 16 rows x 32 cols. Q/K tiles copied with async b128; V transposed
// manually (sVt[dh][key]) so PV B-fragments are contiguous.
// qkv layout: [M=8192, 3*1024], col = which*1024 + h*64 + dh (bf16).
// ---------------------------------------------------------------------------
__global__ __launch_bounds__(256) void attn_kernel(
    const unsigned short* __restrict__ qkv, unsigned short* __restrict__ outb) {
  constexpr int LDT = 72;  // 144B rows
  __shared__ unsigned short sQ[64 * LDT];
  __shared__ unsigned short sK[64 * LDT];
  __shared__ unsigned short sVt[64 * LDT];  // [dh][key]
  __shared__ unsigned short sP[64 * LDT];
  __shared__ float sS[64 * 64];
  __shared__ float rowM[64], rowL[64], rowR[64];

  const int tid = threadIdx.x;
  const int lane = tid & 31, wave = tid >> 5;
  const int wm = wave & 3, wn = wave >> 2;
  const int nl = lane & 15, half = lane >> 4;
  const int q0 = blockIdx.x * 64;
  const int b = blockIdx.y >> 4;
  const int h = blockIdx.y & 15;

  // ---- per-thread staging geometry -------------------------------------
  const int c_r = tid >> 3, c_c = tid & 7;  // b128 chunks: 8 elems along dh
  const unsigned short* gQ =
      qkv + (size_t)(b * NSEQ + q0 + c_r) * 3 * D_MODEL + h * DHEAD + c_c * 8;
  const unsigned short* gK =
      qkv + (size_t)(b * NSEQ + c_r) * 3 * D_MODEL + D_MODEL + h * DHEAD + c_c * 8;
  const unsigned lQ = lds_addr_of(&sQ[c_r * LDT + c_c * 8]);
  const unsigned lK = lds_addr_of(&sK[c_r * LDT + c_c * 8]);
  const unsigned lHalf = 32 * LDT * 2;          // +32 rows, bytes
  const size_t gHalf = (size_t)32 * 3 * D_MODEL;  // +32 rows, elems
  const int v_r = tid >> 5, v_dp = tid & 31;    // V: u32 chunks along dh
  const unsigned int* gV =
      (const unsigned int*)qkv +
      ((size_t)(b * NSEQ + v_r) * 3 * D_MODEL + 2 * D_MODEL + h * DHEAD) / 2 +
      v_dp;
  const int vt0 = (2 * v_dp) * LDT + v_r;       // sVt store bases
  const int vt1 = vt0 + LDT;

  // Q tile (async), row stats init
  async_cp_b128(lQ, gQ);
  async_cp_b128(lQ + lHalf, gQ + gHalf);
  if (tid < 64) {
    rowM[tid] = -1e30f;
    rowL[tid] = 0.f;
  }

  v8f o[2];
#pragma unroll
  for (int t = 0; t < 2; t++)
#pragma unroll
    for (int i = 0; i < 8; i++) o[t][i] = 0.f;

  const int m = wm * 16 + nl;

  for (int kb = 0; kb < NSEQ / 64; kb++) {
    __syncthreads();  // prior-iteration K/V/P consumers done
    // K tile: async; V tile: manual transpose into sVt
    async_cp_b128(lK, gK);
    async_cp_b128(lK + lHalf, gK + gHalf);
#pragma unroll
    for (int it = 0; it < 8; it++) {
      unsigned int vv = gV[(size_t)it * 8 * 3 * D_MODEL / 2];
      sVt[vt0 + it * 8] = (unsigned short)(vv & 0xffffu);
      sVt[vt1 + it * 8] = (unsigned short)(vv >> 16);
    }
    gK += (size_t)64 * 3 * D_MODEL;
    gV += (size_t)64 * 3 * D_MODEL / 2;
    async_wait0();  // covers Q on first iteration too (same counter)
    __syncthreads();

    // S = Q K^T over Dh=64 (two K-steps of 32)
    v8f sacc[2];
#pragma unroll
    for (int t = 0; t < 2; t++)
#pragma unroll
      for (int i = 0; i < 8; i++) sacc[t][i] = 0.f;
#pragma unroll
    for (int ks = 0; ks < 64; ks += 32) {
      const unsigned short* qp = &sQ[m * LDT + ks + 16 * half];
      v16bf aq = frag2(qp, qp + 8);
#pragma unroll
      for (int t = 0; t < 2; t++) {
        int n = wn * 32 + t * 16 + nl;  // key index
        const unsigned short* kp = &sK[n * LDT + ks + 16 * half];
        v16bf bk = frag2(kp, kp + 8);   // contiguous dh run
        sacc[t] = __builtin_amdgcn_wmma_f32_16x16x32_bf16(
            false, aq, false, bk, (short)0, sacc[t], false, false);
      }
    }
    // scale (1/sqrt(64)) and stage S in LDS
#pragma unroll
    for (int t = 0; t < 2; t++)
#pragma unroll
      for (int r = 0; r < 8; r++)
        sS[(wm * 16 + r + 8 * half) * 64 + wn * 32 + t * 16 + nl] =
            sacc[t][r] * 0.125f;
    __syncthreads();

    // Online softmax: waves 0-1 fully active (tid<64 => wave-uniform EXEC)
    if (tid < 64) {
      const int row = tid;
      float mold = rowM[row];
      float mx = mold;
#pragma unroll 8
      for (int c = 0; c < 64; c++) mx = fmaxf(mx, sS[row * 64 + c]);
      float rsc = __expf(mold - mx);
      float l = rowL[row] * rsc;
      float sum = 0.f;
      unsigned int* sPp = (unsigned int*)sP;
#pragma unroll 8
      for (int c = 0; c < 64; c += 2) {
        float p0 = __expf(sS[row * 64 + c] - mx);
        float p1 = __expf(sS[row * 64 + c + 1] - mx);
        sPp[(row * LDT + c) >> 1] =
            (unsigned int)f32_to_bf16(p0) | ((unsigned int)f32_to_bf16(p1) << 16);
        sum += p0 + p1;
      }
      rowL[row] = l + sum;
      rowM[row] = mx;
      rowR[row] = rsc;
    }
    __syncthreads();

    // Rescale O accumulators, then O += P V
#pragma unroll
    for (int t = 0; t < 2; t++)
#pragma unroll
      for (int r = 0; r < 8; r++) o[t][r] *= rowR[wm * 16 + r + 8 * half];

#pragma unroll
    for (int ks = 0; ks < 64; ks += 32) {
      const unsigned short* pp = &sP[m * LDT + ks + 16 * half];
      v16bf ap = frag2(pp, pp + 8);
#pragma unroll
      for (int t = 0; t < 2; t++) {
        int n = wn * 32 + t * 16 + nl;  // dh
        const unsigned short* vp = &sVt[n * LDT + ks + 16 * half];
        v16bf bv = frag2(vp, vp + 8);   // contiguous key run
        o[t] = __builtin_amdgcn_wmma_f32_16x16x32_bf16(
            false, ap, false, bv, (short)0, o[t], false, false);
      }
    }
  }
  __syncthreads();

  // Write context (bf16) laid out as [M, D] with col = h*64 + dh
#pragma unroll
  for (int t = 0; t < 2; t++) {
    int dcol = h * DHEAD + wn * 32 + t * 16 + nl;
#pragma unroll
    for (int r = 0; r < 8; r++) {
      int rl = wm * 16 + r + 8 * half;
      float v = o[t][r] / rowL[rl];
      outb[(size_t)(b * NSEQ + q0 + rl) * D_MODEL + dcol] = f32_to_bf16(v);
    }
  }
}

// ---------------------------------------------------------------------------
// Host orchestration
// ---------------------------------------------------------------------------
extern "C" void kernel_launch(void* const* d_in, const int* in_sizes, int n_in,
                              void* d_out, int out_size, void* d_ws,
                              size_t ws_size, hipStream_t stream) {
  (void)in_sizes; (void)n_in; (void)out_size; (void)ws_size;
  const float* x      = (const float*)d_in[0];
  const float* ln1_g  = (const float*)d_in[1];
  const float* ln1_b  = (const float*)d_in[2];
  const float* qkv_w  = (const float*)d_in[3];
  const float* qkv_b  = (const float*)d_in[4];
  const float* proj_w = (const float*)d_in[5];
  const float* proj_b = (const float*)d_in[6];
  const float* ln2_g  = (const float*)d_in[7];
  const float* ln2_b  = (const float*)d_in[8];
  const float* fc1_w  = (const float*)d_in[9];
  const float* fc1_b  = (const float*)d_in[10];
  const float* fc2_w  = (const float*)d_in[11];
  const float* fc2_b  = (const float*)d_in[12];
  float* out = (float*)d_out;

  char* ws = (char*)d_ws;
  size_t off = 0;
  auto carve = [&](size_t bytes) {
    void* p = ws + off;
    off += (bytes + 255) & ~(size_t)255;
    return p;
  };
  unsigned short* wqkv  = (unsigned short*)carve((size_t)D_MODEL * 3 * D_MODEL * 2);
  unsigned short* wproj = (unsigned short*)carve((size_t)D_MODEL * D_MODEL * 2);
  unsigned short* wfc1  = (unsigned short*)carve((size_t)D_MODEL * HIDDEN * 2);
  unsigned short* wfc2  = (unsigned short*)carve((size_t)HIDDEN * D_MODEL * 2);
  unsigned short* h1    = (unsigned short*)carve((size_t)MTOT * D_MODEL * 2);
  unsigned short* qkvb  = (unsigned short*)carve((size_t)MTOT * 3 * D_MODEL * 2);
  unsigned short* attn  = (unsigned short*)carve((size_t)MTOT * D_MODEL * 2);
  float*          x1    = (float*)carve((size_t)MTOT * D_MODEL * 4);
  unsigned short* h2    = (unsigned short*)carve((size_t)MTOT * D_MODEL * 2);
  unsigned short* g1    = (unsigned short*)carve((size_t)MTOT * HIDDEN * 2);

  // Weight converts (fp32 -> bf16)
  cvt_f32_bf16<<<2048, 256, 0, stream>>>(qkv_w, wqkv, D_MODEL * 3 * D_MODEL);
  cvt_f32_bf16<<<2048, 256, 0, stream>>>(proj_w, wproj, D_MODEL * D_MODEL);
  cvt_f32_bf16<<<2048, 256, 0, stream>>>(fc1_w, wfc1, D_MODEL * HIDDEN);
  cvt_f32_bf16<<<2048, 256, 0, stream>>>(fc2_w, wfc2, HIDDEN * D_MODEL);

  // LN1
  ln_bf16_kernel<<<MTOT, 256, 0, stream>>>(x, ln1_g, ln1_b, h1);
  // QKV GEMM: [8192,1024]x[1024,3072] -> bf16
  gemm_bf16<0><<<dim3(3 * D_MODEL / 64, MTOT / 128), 256, 0, stream>>>(
      h1, wqkv, qkv_b, nullptr, nullptr, qkvb, MTOT, 3 * D_MODEL, D_MODEL);
  // Attention (flash, per b,h, 64-row query tiles)
  attn_kernel<<<dim3(NSEQ / 64, BATCH * NHEAD), 256, 0, stream>>>(qkvb, attn);
  // Proj GEMM + residual(x) -> x1 (fp32)
  gemm_bf16<2><<<dim3(D_MODEL / 64, MTOT / 128), 256, 0, stream>>>(
      attn, wproj, proj_b, x, x1, nullptr, MTOT, D_MODEL, D_MODEL);
  // LN2
  ln_bf16_kernel<<<MTOT, 256, 0, stream>>>(x1, ln2_g, ln2_b, h2);
  // FC1 GEMM + exact GELU -> bf16
  gemm_bf16<1><<<dim3(HIDDEN / 64, MTOT / 128), 256, 0, stream>>>(
      h2, wfc1, fc1_b, nullptr, nullptr, g1, MTOT, HIDDEN, D_MODEL);
  // FC2 GEMM + residual(x1) -> d_out (fp32)
  gemm_bf16<2><<<dim3(D_MODEL / 64, MTOT / 128), 256, 0, stream>>>(
      g1, wfc2, fc2_b, x1, out, nullptr, MTOT, D_MODEL, HIDDEN);
}
